// VQ1D_8658654069376
// MI455X (gfx1250) — matrix operand order repaired and checked by reading, not compiled, not run on hardware
//
#include <hip/hip_runtime.h>
#include <hip/hip_bf16.h>

// VQ nearest-codebook search on gfx1250 (MI455X).
// B=16, L=2048, D=128, K=8192, 1 RQ step.
// d_out = [z_q (16*2048*128 f32)] ++ [indices as f32 (16*2048)]

#define DIMS     128
#define KCODES   8192
#define BL       32768          // B*L rows
#define ZQ_ELEMS (BL * DIMS)    // 4194304
#define NT_CHUNK 64             // codes staged in LDS per phase

typedef __attribute__((ext_vector_type(16))) __bf16 v16bf;
typedef __attribute__((ext_vector_type(8)))  float  v8f;

// ---------------------------------------------------------------------------
// Prep: split codebook f32 -> bf16 hi/lo (compensated), compute ||c||^2 (f32).
// One block (128 threads) per code.
// ---------------------------------------------------------------------------
__global__ __launch_bounds__(128) void vq_prep(const float* __restrict__ cb,
                                               __bf16* __restrict__ cb_hi,
                                               __bf16* __restrict__ cb_lo,
                                               float* __restrict__ cb_norm) {
  const int k = blockIdx.x;
  const int t = threadIdx.x;
  const float x = cb[k * DIMS + t];
  const __bf16 h = (__bf16)x;
  cb_hi[k * DIMS + t] = h;
  cb_lo[k * DIMS + t] = (__bf16)(x - (float)h);

  float s = x * x;
  #pragma unroll
  for (int o = 16; o >= 1; o >>= 1) s += __shfl_down(s, o, 32);
  __shared__ float part[4];
  if ((t & 31) == 0) part[t >> 5] = s;
  __syncthreads();
  if (t == 0) cb_norm[k] = part[0] + part[1] + part[2] + part[3];
}

// ---------------------------------------------------------------------------
// Main: each wave owns a 16-row query tile, loops all 8192 codes.
// 4 waves/block share 64-code LDS chunks of the bf16 codebook.
// Per 16x16 tile: 4 k-steps x 3 compensated WMMAs (hi*hi + hi*lo + lo*hi).
// Fused epilogue: d2 = ||c||^2 - 2*dot, running argmin, xor-shuffle reduce.
// ---------------------------------------------------------------------------
__global__ __launch_bounds__(128) void vq_main(const float* __restrict__ ze,
                                               const __bf16* __restrict__ cb_hi,
                                               const __bf16* __restrict__ cb_lo,
                                               const float* __restrict__ cb_norm,
                                               float* __restrict__ out_idx) {
  __shared__ __align__(16) __bf16 sm_hi[NT_CHUNK * DIMS];  // 16 KB
  __shared__ __align__(16) __bf16 sm_lo[NT_CHUNK * DIMS];  // 16 KB

  const int lane = threadIdx.x & 31;
  const int wave = threadIdx.x >> 5;
  const int m0   = (blockIdx.x * 4 + wave) * 16;   // query tile base row
  const int half = lane >> 4;                      // 0 / 1 (K-half owner)
  const int l15  = lane & 15;

  // ---- Build A fragments (16 queries x 128 dims) in registers, hi/lo split.
  // A 16x32 layout (contiguous, matching B): lanes row=l15 hold K = half*16 + e.
  // base = kk*32 + half*16.
  v16bf a_hi[4], a_lo[4];
  {
    const float* arow = ze + (size_t)(m0 + l15) * DIMS;
    #pragma unroll
    for (int kk = 0; kk < 4; ++kk) {
      const int base = kk * 32 + half * 16;
      const float4 p0 = *(const float4*)(arow + base);
      const float4 p1 = *(const float4*)(arow + base + 4);
      const float4 p2 = *(const float4*)(arow + base + 8);
      const float4 p3 = *(const float4*)(arow + base + 12);
      const float v[16] = {p0.x, p0.y, p0.z, p0.w, p1.x, p1.y, p1.z, p1.w,
                           p2.x, p2.y, p2.z, p2.w, p3.x, p3.y, p3.z, p3.w};
      #pragma unroll
      for (int e = 0; e < 16; ++e) {
        const __bf16 h = (__bf16)v[e];
        a_hi[kk][e] = h;
        a_lo[kk][e] = (__bf16)(v[e] - (float)h);
      }
    }
  }

  float minv[8];
  int   mini[8];
  #pragma unroll
  for (int r = 0; r < 8; ++r) { minv[r] = 3.4e38f; mini[r] = 0; }

  const int dofs = half * 16;  // B layout: high lanes own K 16..31 of a k-step

  for (int nc = 0; nc < KCODES; nc += NT_CHUNK) {
    __syncthreads();
    // Stage 64 codes (hi+lo) into LDS: 128 threads x 8 float4 per matrix.
    {
      const float4* gh = (const float4*)(cb_hi + (size_t)nc * DIMS);
      const float4* gl = (const float4*)(cb_lo + (size_t)nc * DIMS);
      float4* sh = (float4*)sm_hi;
      float4* sl = (float4*)sm_lo;
      #pragma unroll
      for (int i = 0; i < 8; ++i) {
        sh[threadIdx.x + i * 128] = gh[threadIdx.x + i * 128];
        sl[threadIdx.x + i * 128] = gl[threadIdx.x + i * 128];
      }
      if (nc + NT_CHUNK < KCODES) {  // warm next chunk (global_prefetch_b8)
        __builtin_prefetch(cb_hi + (size_t)(nc + NT_CHUNK) * DIMS + threadIdx.x * 64, 0, 1);
        __builtin_prefetch(cb_lo + (size_t)(nc + NT_CHUNK) * DIMS + threadIdx.x * 64, 0, 1);
      }
    }
    __syncthreads();

    #pragma unroll
    for (int t4 = 0; t4 < 4; ++t4) {            // 4 n-tiles of 16 codes
      const int n0  = nc + t4 * 16;
      const int col = t4 * 16 + l15;            // local code within chunk
      v8f acc = {};
      #pragma unroll
      for (int kk = 0; kk < 4; ++kk) {          // 128 dims = 4 x 32
        const v16bf bh = *(const v16bf*)(sm_hi + col * DIMS + kk * 32 + dofs);
        const v16bf bl = *(const v16bf*)(sm_lo + col * DIMS + kk * 32 + dofs);
        acc = __builtin_amdgcn_wmma_f32_16x16x32_bf16(false, a_hi[kk], false, bh,
                                                      (short)0, acc, false, false);
        acc = __builtin_amdgcn_wmma_f32_16x16x32_bf16(false, a_hi[kk], false, bl,
                                                      (short)0, acc, false, false);
        acc = __builtin_amdgcn_wmma_f32_16x16x32_bf16(false, a_lo[kk], false, bh,
                                                      (short)0, acc, false, false);
      }
      // d2 = ||c||^2 - 2*dot  (||x||^2 is constant per row -> drop for argmin)
      const float cn   = cb_norm[n0 + l15];
      const int   code = n0 + l15;
      #pragma unroll
      for (int r = 0; r < 8; ++r) {
        const float d2 = fmaf(-2.0f, acc[r], cn);
        if (d2 < minv[r]) { minv[r] = d2; mini[r] = code; }   // strict < keeps
      }                                                       // first occurrence
    }
  }

  // Argmin across the 16 lanes of each half (xor masks < 16 stay in-half).
  // C/D layout: VGPR r holds rows M=r (lanes 0-15) and M=r+8 (lanes 16-31).
  #pragma unroll
  for (int r = 0; r < 8; ++r) {
    float v = minv[r];
    int   i = mini[r];
    #pragma unroll
    for (int off = 8; off >= 1; off >>= 1) {
      const float ov = __shfl_xor(v, off, 32);
      const int   oi = __shfl_xor(i, off, 32);
      if (ov < v || (ov == v && oi < i)) { v = ov; i = oi; }  // jnp.argmin ties
    }
    if (l15 == 0) out_idx[m0 + half * 8 + r] = (float)i;
  }
}

// ---------------------------------------------------------------------------
// Gather: z_q[row] = codebook[idx[row]]   (one block per row, 128 threads)
// ---------------------------------------------------------------------------
__global__ __launch_bounds__(128) void vq_gather(const float* __restrict__ idxf,
                                                 const float* __restrict__ cb,
                                                 float* __restrict__ zq) {
  const int row = blockIdx.x;
  const int t   = threadIdx.x;
  const int idx = (int)idxf[row];
  zq[(size_t)row * DIMS + t] = cb[(size_t)idx * DIMS + t];
}

// ---------------------------------------------------------------------------
extern "C" void kernel_launch(void* const* d_in, const int* in_sizes, int n_in,
                              void* d_out, int out_size, void* d_ws, size_t ws_size,
                              hipStream_t stream) {
  (void)in_sizes; (void)n_in; (void)out_size; (void)ws_size;
  const float* ze = (const float*)d_in[0];   // [B,L,D] f32
  const float* cb = (const float*)d_in[1];   // [K,D]   f32

  float* zq   = (float*)d_out;               // [B,L,D]
  float* oidx = (float*)d_out + ZQ_ELEMS;    // [B,L,1] as f32

  // Workspace: cb_hi (2MB) | cb_lo (2MB) | cb_norm (32KB)
  __bf16* cb_hi   = (__bf16*)d_ws;
  __bf16* cb_lo   = cb_hi + (size_t)KCODES * DIMS;
  float*  cb_norm = (float*)(cb_lo + (size_t)KCODES * DIMS);

  vq_prep  <<<KCODES, 128, 0, stream>>>(cb, cb_hi, cb_lo, cb_norm);
  vq_main  <<<BL / 16 / 4, 128, 0, stream>>>(ze, cb_hi, cb_lo, cb_norm, oidx);
  vq_gather<<<BL, 128, 0, stream>>>(oidx, cb, zq);
}
